// MultiHeadAttention_78469052498460
// MI455X (gfx1250) — compile-verified, matrix operand order checked by reading, not run on hardware
//
#include <hip/hip_runtime.h>
#include <stdint.h>

// ---------------------------------------------------------------------------
// MultiHeadAttention forward for MI455X (gfx1250, wave32, WMMA).
//
//  reference:  q,k,v = x@W{q,k,v}^T ; logits = (q k^T)/sqrt(64), masked with
//              -1e18 ; attention_weights = sum_h(masked logits)/8 ;
//              atten = softmax(logits) @ v   (heads concat on feature dim).
//
//  Roofline: ~38 GFLOP vs ~100 MB mandatory HBM traffic (64 MB of it the
//  [4096,4096] attention_weights write) -> memory-bound, ~4.3us floor at
//  23.3 TB/s.  Matmuls run on v_wmma_f32_16x16x32_f16 (f16 in, f32 acc).
//
//  Fragment I/O exploits the ISA 7.12.2 layout: a 16-bit 16x32 A/B fragment
//  is, per lane, two contiguous 16-byte runs (k = kb0..kb0+7 and
//  16+kb0..16+kb0+7, kb0 = lane>=16 ? 8 : 0) -> 2x b128 loads per fragment
//  instead of 16 scalar half loads.
//
//  Pass 1 (proj_kernel):  Q,K (row-major f16) and V^T (f16 [512][4096]) into
//                         workspace; one wave per 16x16 tile, 16 WMMAs.
//  Pass 2 (attn_kernel):  1 block = 16 query rows, wave w = head w.
//                         Flash-style online softmax over 64-key blocks;
//                         attention_weights = sum over the 8 per-head logit
//                         tiles in LDS at write-out (no atomics); P.V via
//                         WMMA against V^T.  K/V for the next key block are
//                         prefetched (global_prefetch_b8); no TDM staging --
//                         each wave reads a distinct head slice (no intra-
//                         block reuse) and L2 (192MB) captures cross-block
//                         reuse of the 8MB K/V working set.
//
//  Workspace: 3 * 4096*512 * 2B = 12 MiB (Qh | Kh | Vt).
// ---------------------------------------------------------------------------

typedef _Float16 half_t;
typedef __attribute__((ext_vector_type(16))) _Float16 v16h;
typedef __attribute__((ext_vector_type(8)))  _Float16 v8h;
typedef __attribute__((ext_vector_type(4)))  _Float16 v4h;
typedef __attribute__((ext_vector_type(8)))  float    v8f;
typedef __attribute__((ext_vector_type(4)))  float    v4f;

#define S_LEN  4096
#define EMB    512
#define NHEAD  8
#define DHEAD  64
#define BK     64           // keys per online-softmax block
#define MASKF  (-1.0e18f)

// C/D fragment row for VGPR r (ISA 7.12.2: VGPR r holds M=r (lanes<16), r+8).
__device__ __forceinline__ int crow(int lane, int r) {
  return r + ((lane & 16) >> 1);
}

// f16 16x32 A/B fragment from a row-major row: p = &M[row][k0], 16B-aligned.
__device__ __forceinline__ v16h ld_frag_f16(const half_t* __restrict__ p,
                                            int lane) {
  const int kb0 = (lane & 16) >> 1;                 // 0 | 8
  v8h lo = *(const v8h*)(p + kb0);                  // k = kb0 .. kb0+7
  v8h hi = *(const v8h*)(p + 16 + kb0);             // k = 16+kb0 .. 16+kb0+7
  return __builtin_shufflevector(lo, hi, 0, 1, 2, 3, 4, 5, 6, 7,
                                 8, 9, 10, 11, 12, 13, 14, 15);
}

// Same fragment sourced from f32 memory (projection inputs), with convert.
__device__ __forceinline__ v16h ld_frag_f32(const float* __restrict__ p,
                                            int lane) {
  const int kb0 = (lane & 16) >> 1;
  v4f a = *(const v4f*)(p + kb0);
  v4f b = *(const v4f*)(p + kb0 + 4);
  v4f c = *(const v4f*)(p + 16 + kb0);
  v4f d = *(const v4f*)(p + 16 + kb0 + 4);
  v16h f;
#pragma unroll
  for (int j = 0; j < 4; ++j) {
    f[j]      = (half_t)a[j];
    f[j + 4]  = (half_t)b[j];
    f[j + 8]  = (half_t)c[j];
    f[j + 12] = (half_t)d[j];
  }
  return f;
}

__device__ __forceinline__ v8f wmma_f16(v16h a, v16h b, v8f c) {
  return __builtin_amdgcn_wmma_f32_16x16x32_f16(false, a, false, b,
                                                (short)0, c, false, false);
}

// ---------------------------------------------------------------------------
// Pass 1: projections.  grid = (EMB/128, S/16, 3), block = 256 (8 waves).
// Wave w computes the 16x16 tile at n = (bx*8+w)*16, m = by*16 of x @ W^T.
// ---------------------------------------------------------------------------
__global__ __launch_bounds__(256)
void proj_kernel(const float* __restrict__ x,
                 const float* __restrict__ Wq,
                 const float* __restrict__ Wk,
                 const float* __restrict__ Wv,
                 half_t* __restrict__ Qh,
                 half_t* __restrict__ Kh,
                 half_t* __restrict__ Vt) {
  const int lane  = threadIdx.x & 31;
  const int wv    = threadIdx.x >> 5;
  const int nbase = (blockIdx.x * 8 + wv) * 16;
  const int mbase = blockIdx.y * 16;
  const int mat   = blockIdx.z;                      // 0=Q 1=K 2=V
  const float* __restrict__ W = (mat == 0) ? Wq : (mat == 1) ? Wk : Wv;

  const float* xrow = &x[(size_t)(mbase + (lane & 15)) * EMB];  // A rows
  const float* wrow = &W[(size_t)(nbase + (lane & 15)) * EMB];  // B "rows"

  v8f acc = {};
  for (int ks = 0; ks < EMB; ks += 32) {
    v16h a = ld_frag_f32(xrow + ks, lane);           // A: x[m][k]
    v16h b = ld_frag_f32(wrow + ks, lane);           // B: W[n][k] = (W^T)[k][n]
    acc = wmma_f16(a, b, acc);
  }

  if (mat == 2) {                                    // V stored transposed:
    v8h pk;                                          // 8 consecutive s per lane
#pragma unroll
    for (int r = 0; r < 8; ++r) pk[r] = (half_t)acc[r];
    *(v8h*)&Vt[(size_t)(nbase + (lane & 15)) * S_LEN + mbase +
               ((lane & 16) >> 1)] = pk;
  } else {
    half_t* __restrict__ dst = (mat == 0) ? Qh : Kh;
#pragma unroll
    for (int r = 0; r < 8; ++r)
      dst[(size_t)(mbase + crow(lane, r)) * EMB + nbase + (lane & 15)] =
          (half_t)acc[r];
  }
}

// ---------------------------------------------------------------------------
// Pass 2: attention.  grid = S/16 blocks, block = 256 threads = 8 waves.
// Wave h handles head h for the block's 16 query rows (flash-style).
// ---------------------------------------------------------------------------
__global__ __launch_bounds__(256)
void attn_kernel(const half_t* __restrict__ Qh,
                 const half_t* __restrict__ Kh,
                 const half_t* __restrict__ Vt,
                 const uint8_t* __restrict__ mask,
                 float* __restrict__ atten,      // [S][EMB]
                 float* __restrict__ aw) {       // [S][S]
  __shared__ float  s_S[NHEAD][16][BK];          // masked scaled logits (f32)
  __shared__ half_t s_P[NHEAD][16][BK];          // exp(logit - m), f16 A-src
  __shared__ float  s_m[NHEAD][16];              // running row max
  __shared__ float  s_l[NHEAD][16];              // running row sum
  __shared__ float  s_a[NHEAD][16];              // per-step rescale alpha

  const int lane  = threadIdx.x & 31;
  const int h     = threadIdx.x >> 5;            // wave == head
  const int qbase = blockIdx.x * 16;
  const int ar    = lane & 15;                   // A/B fragment row-in-tile
  const float scale = 0.125f;                    // 1/sqrt(DHEAD)

  // Q fragments for this head: 16 x 64 = two 16x32 A fragments, resident.
  v16h qf[2];
#pragma unroll
  for (int ks = 0; ks < 2; ++ks)
    qf[ks] = ld_frag_f16(&Qh[(size_t)(qbase + ar) * EMB + h * DHEAD + ks * 32],
                         lane);

  v8f o[4] = {};                                 // 16 x 64 f32 accumulator

  if (lane < 16) { s_m[h][lane] = -3.0e38f; s_l[h][lane] = 0.0f; }

  for (int kb = 0; kb < S_LEN; kb += BK) {
    __syncthreads();   // previous block's s_S fully consumed by aw writer

    if (kb + BK < S_LEN) {                       // global_prefetch_b8
      __builtin_prefetch(&Kh[(size_t)(kb + BK + ar) * EMB + h * DHEAD], 0, 1);
      __builtin_prefetch(&Vt[(size_t)(h * DHEAD + ar) * S_LEN + kb + BK], 0, 1);
    }

    // ---- logits: 4 key tiles of 16, each = 2 WMMAs over dk=64 -------------
#pragma unroll
    for (int kt = 0; kt < 4; ++kt) {
      const int kcol = kb + kt * 16;
      const half_t* krow = &Kh[(size_t)(kcol + ar) * EMB + h * DHEAD];
      v16h b0 = ld_frag_f16(krow, lane);         // B: K[n][k]  (Q K^T)
      v16h b1 = ld_frag_f16(krow + 32, lane);
      v8f c = {};
      c = wmma_f16(qf[0], b0, c);
      c = wmma_f16(qf[1], b1, c);
#pragma unroll
      for (int r = 0; r < 8; ++r) {
        const int m = crow(lane, r);
        float sv = c[r] * scale;
        if (mask[(size_t)(qbase + m) * S_LEN + kcol + ar]) sv = MASKF;
        s_S[h][m][kt * 16 + ar] = sv;
      }
    }

    // ---- online softmax update (wave-local; 2 lanes per row) --------------
    {
      const int row = ar;
      const int hs  = lane >> 4;                 // which half of the 64 cols
      const float* srow = &s_S[h][row][hs * 32];
      half_t*      prow = &s_P[h][row][hs * 32];
      const float mold = s_m[h][row];
      const float lold = s_l[h][row];
      float mx = -3.0e38f;
#pragma unroll
      for (int c4 = 0; c4 < 8; ++c4) {
        v4f sv = *(const v4f*)(srow + 4 * c4);
        mx = fmaxf(mx, fmaxf(fmaxf(sv[0], sv[1]), fmaxf(sv[2], sv[3])));
      }
      mx = fmaxf(mx, __shfl_xor(mx, 16, 32));
      const float mnew  = fmaxf(mx, mold);
      const float alpha = __expf(mold - mnew);
      float psum = 0.0f;
#pragma unroll
      for (int c4 = 0; c4 < 8; ++c4) {
        v4f sv = *(const v4f*)(srow + 4 * c4);
        v4h pk;
#pragma unroll
        for (int j = 0; j < 4; ++j) {
          const float p = __expf(sv[j] - mnew);
          psum += p;
          pk[j] = (half_t)p;
        }
        *(v4h*)(prow + 4 * c4) = pk;             // ds_store_b64
      }
      psum += __shfl_xor(psum, 16, 32);
      if (hs == 0) {
        s_m[h][row] = mnew;
        s_l[h][row] = lold * alpha + psum;
        s_a[h][row] = alpha;
      }
    }

    // ---- rescale accumulator, then P.V (8 WMMAs against V^T) --------------
#pragma unroll
    for (int nt = 0; nt < 4; ++nt)
#pragma unroll
      for (int r = 0; r < 8; ++r)
        o[nt][r] *= s_a[h][crow(lane, r)];

#pragma unroll
    for (int ks = 0; ks < 2; ++ks) {
      v16h pa = ld_frag_f16(&s_P[h][ar][ks * 32], lane);   // ds_load_b128 x2
#pragma unroll
      for (int nt = 0; nt < 4; ++nt) {
        v16h vb = ld_frag_f16(
            &Vt[(size_t)(h * DHEAD + nt * 16 + ar) * S_LEN + kb + ks * 32],
            lane);                               // B: V^T rows contiguous in s
        o[nt] = wmma_f16(pa, vb, o[nt]);
      }
    }

    __syncthreads();   // all 8 heads' s_S tiles visible
    // ---- stream out attention_weights block (the dominant HBM traffic) ----
    for (int e = threadIdx.x; e < 16 * BK; e += 256) {
      const int m = e >> 6, c = e & 63;
      float accw = 0.0f;
#pragma unroll
      for (int hh = 0; hh < NHEAD; ++hh) accw += s_S[hh][m][c];
      aw[(size_t)(qbase + m) * S_LEN + kb + c] = accw * (1.0f / NHEAD);
    }
  }

  // ---- epilogue: normalize and write this head's 16x64 output tile --------
#pragma unroll
  for (int nt = 0; nt < 4; ++nt)
#pragma unroll
    for (int r = 0; r < 8; ++r) {
      const int m = crow(lane, r);
      atten[(size_t)(qbase + m) * EMB + h * DHEAD + nt * 16 + ar] =
          o[nt][r] * (1.0f / s_l[h][m]);
    }
}

// ---------------------------------------------------------------------------
extern "C" void kernel_launch(void* const* d_in, const int* in_sizes, int n_in,
                              void* d_out, int out_size, void* d_ws,
                              size_t ws_size, hipStream_t stream) {
  (void)in_sizes; (void)n_in; (void)out_size; (void)ws_size;
  const float*   x    = (const float*)d_in[0];
  const uint8_t* mask = (const uint8_t*)d_in[1];   // jnp.bool_ -> 1 byte/elem
  const float*   Wq   = (const float*)d_in[2];
  const float*   Wk   = (const float*)d_in[3];
  const float*   Wv   = (const float*)d_in[4];

  float* atten = (float*)d_out;                          // [4096][512]
  float* aw    = (float*)d_out + (size_t)S_LEN * EMB;    // [4096][4096]

  half_t* Qh = (half_t*)d_ws;                            // 4 MiB
  half_t* Kh = Qh + (size_t)S_LEN * EMB;                 // 4 MiB
  half_t* Vt = Kh + (size_t)S_LEN * EMB;                 // 4 MiB, [EMB][S]

  proj_kernel<<<dim3(EMB / 128, S_LEN / 16, 3), 256, 0, stream>>>(
      x, Wq, Wk, Wv, Qh, Kh, Vt);
  attn_kernel<<<dim3(S_LEN / 16), 256, 0, stream>>>(
      Qh, Kh, Vt, mask, atten, aw);
}
